// MultiHeadAttention_68642167324853
// MI455X (gfx1250) — compile-verified, matrix operand order checked by reading, not run on hardware
//
#include <hip/hip_runtime.h>

// ---------------------------------------------------------------------------
// MHA forward for B=4, S=2048, D=1024, H=16, DH=64 on gfx1250 (MI455X).
// bf16 WMMA (f32 accumulate) for all GEMMs + flash attention.
// GEMM tiles are double-buffered in LDS and staged with
// GLOBAL_LOAD_ASYNC_TO_LDS_B128 (ASYNCcnt-tracked), overlapping the next
// K-slab's HBM traffic with the current slab's WMMAs.
// ---------------------------------------------------------------------------

typedef __attribute__((ext_vector_type(16))) __bf16 bf16x16;
typedef __attribute__((ext_vector_type(8)))  __bf16 bf16x8;
typedef __attribute__((ext_vector_type(4)))  __bf16 bf16x4;
typedef __attribute__((ext_vector_type(8)))  float  f32x8;
typedef __attribute__((ext_vector_type(4)))  float  f32x4;
typedef __attribute__((ext_vector_type(4)))  int    i32x4;
typedef int v4i __attribute__((vector_size(16)));   // matches builtin's V4i

union FragBF { bf16x16 v; bf16x8 h[2]; };

#define WMMA_BF16(a, b, c) \
  __builtin_amdgcn_wmma_f32_16x16x32_bf16(false, (a), false, (b), (short)0, (c), false, false)

#if defined(__has_builtin)
#if __has_builtin(__builtin_amdgcn_global_load_async_to_lds_b128) && \
    __has_builtin(__builtin_amdgcn_s_wait_asynccnt)
#define HAVE_ASYNC_LDS 1
#endif
#endif
#ifndef HAVE_ASYNC_LDS
#define HAVE_ASYNC_LDS 0
#endif

// Copy 16 bytes global -> LDS. Async (ASYNCcnt-tracked, writes LDS directly)
// when available; otherwise a b128 load + ds_store.
__device__ __forceinline__ void cp_lds16(void* lds, const void* gmem) {
#if HAVE_ASYNC_LDS
  __builtin_amdgcn_global_load_async_to_lds_b128(
      (__attribute__((address_space(1))) v4i*)gmem,
      (__attribute__((address_space(3))) v4i*)lds, 0, 0);
#else
  *reinterpret_cast<i32x4*>(lds) = *reinterpret_cast<const i32x4*>(gmem);
#endif
}

template <int N>
__device__ __forceinline__ void cp_lds_wait() {
#if HAVE_ASYNC_LDS
  __builtin_amdgcn_s_wait_asynccnt(N);
#endif
}

// ---------------------------------------------------------------------------
// fp32 -> bf16 elementwise (n4 = n/4 vector groups)
// ---------------------------------------------------------------------------
__global__ __launch_bounds__(256)
void k_cvt_bf16(const float* __restrict__ in, __bf16* __restrict__ out, int n4) {
  int i = blockIdx.x * 256 + threadIdx.x;
  if (i >= n4) return;
  f32x4 v = reinterpret_cast<const f32x4*>(in)[i];
  bf16x4 o;
#pragma unroll
  for (int j = 0; j < 4; ++j) o[j] = (__bf16)v[j];
  reinterpret_cast<bf16x4*>(out)[i] = o;
}

// ---------------------------------------------------------------------------
// W fp32 [K][N] -> Wt bf16 [N][K]  (so B-operand loads are contiguous in K)
// ---------------------------------------------------------------------------
__global__ __launch_bounds__(256)
void k_transpose_bf16(const float* __restrict__ W, __bf16* __restrict__ Wt,
                      int K, int N) {
  __shared__ float tile[32][33];
  const int tx = threadIdx.x, ty = threadIdx.y;
  const int n0 = blockIdx.x * 32, k0 = blockIdx.y * 32;
#pragma unroll
  for (int i = 0; i < 4; ++i)
    tile[ty + i * 8][tx] = W[(size_t)(k0 + ty + i * 8) * N + n0 + tx];
  __syncthreads();
#pragma unroll
  for (int i = 0; i < 4; ++i)
    Wt[(size_t)(n0 + ty + i * 8) * K + k0 + tx] = (__bf16)tile[tx][ty + i * 8];
}

// ---------------------------------------------------------------------------
// GEMM: C[M][N] = A[M][K] * Bt[N][K]^T + bias[N]
//   mode 0: fp32 row-major to outF;  mode 1: bf16 to outH in [B,H,S,DH]
// 128x128 block tile, 8 waves (2 M x 4 N), wave = 64x32 = 4x2 WMMA tiles.
// K staged 64-deep, double-buffered in LDS via async loads.
// ---------------------------------------------------------------------------
__global__ __launch_bounds__(256)
void k_gemm_bf16(const __bf16* __restrict__ A, const __bf16* __restrict__ Bt,
                 const float* __restrict__ bias, float* __restrict__ outF,
                 __bf16* __restrict__ outH, int M, int N, int K,
                 int mode, int S, int Hn, int DHn) {
  __shared__ __align__(16) __bf16 As[2][128][64];
  __shared__ __align__(16) __bf16 Bs[2][128][64];

  const int tid  = threadIdx.x;
  const int wid  = tid >> 5;
  const int lane = tid & 31;
  const int half = lane >> 4;
  const int l16  = lane & 15;
  const int bm = blockIdx.x * 128;
  const int bn = blockIdx.y * 128;
  const int wave_m = (wid & 1) * 64;   // 2 waves along M
  const int wave_n = (wid >> 1) * 32;  // 4 waves along N

  f32x8 acc[4][2];
  const f32x8 vzero = {};
#pragma unroll
  for (int tm = 0; tm < 4; ++tm)
#pragma unroll
    for (int tn = 0; tn < 2; ++tn) acc[tm][tn] = vzero;

  const int lr = tid >> 1;  // 0..127 LDS row
  const int lc = tid & 1;   // 0/1: 32-element segment

  // Issue one 128x64 A tile + 128x64 B tile into buffer `buf`.
  // 4+4 async b128 per thread => 8 outstanding ASYNCcnt per wave.
  auto issue_tile = [&](int kk, int buf) {
#pragma unroll
    for (int i = 0; i < 4; ++i)
      cp_lds16(&As[buf][lr][lc * 32 + i * 8],
               &A[(size_t)(bm + lr) * K + kk + lc * 32 + i * 8]);
#pragma unroll
    for (int i = 0; i < 4; ++i)
      cp_lds16(&Bs[buf][lr][lc * 32 + i * 8],
               &Bt[(size_t)(bn + lr) * K + kk + lc * 32 + i * 8]);
  };

  issue_tile(0, 0);
  for (int kk = 0; kk < K; kk += 64) {
    const int buf = (kk >> 6) & 1;
    if (kk + 64 < K) {
      issue_tile(kk + 64, buf ^ 1);   // prefetch next slab into other buffer
      cp_lds_wait<8>();               // current slab (older 8 ops) has landed
      if (kk + 128 < K)
        __builtin_prefetch(&A[(size_t)(bm + lr) * K + kk + 128 + lc * 32], 0, 1);
    } else {
      cp_lds_wait<0>();
    }
    __syncthreads();

#pragma unroll
    for (int ks = 0; ks < 64; ks += 32) {
      FragBF bfr[2];
#pragma unroll
      for (int tn = 0; tn < 2; ++tn) {
        const int n = wave_n + tn * 16 + l16;
        bfr[tn].h[0] = *reinterpret_cast<const bf16x8*>(&Bs[buf][n][ks + half * 8]);
        bfr[tn].h[1] = *reinterpret_cast<const bf16x8*>(&Bs[buf][n][ks + 16 + half * 8]);
      }
#pragma unroll
      for (int tm = 0; tm < 4; ++tm) {
        FragBF afr;
        const int m = wave_m + tm * 16 + l16;
        afr.h[0] = *reinterpret_cast<const bf16x8*>(&As[buf][m][ks + half * 8]);
        afr.h[1] = *reinterpret_cast<const bf16x8*>(&As[buf][m][ks + 16 + half * 8]);
#pragma unroll
        for (int tn = 0; tn < 2; ++tn)
          acc[tm][tn] = WMMA_BF16(afr.v, bfr[tn].v, acc[tm][tn]);
      }
    }
    __syncthreads();  // reads of `buf` done before it is refilled next+1 iter
  }

  // Epilogue. C layout: VGPR e -> row M = e + 8*half, col N = lane%16.
#pragma unroll
  for (int tm = 0; tm < 4; ++tm) {
#pragma unroll
    for (int tn = 0; tn < 2; ++tn) {
      const int n = bn + wave_n + tn * 16 + l16;
      const float bv = bias[n];
      const int m0 = bm + wave_m + tm * 16;
#pragma unroll
      for (int e = 0; e < 8; ++e) {
        const int m = m0 + e + half * 8;
        const float val = acc[tm][tn][e] + bv;
        if (mode == 0) {
          outF[(size_t)m * N + n] = val;
        } else {
          const int b = m / S, s = m % S;
          const int h = n / DHn, dh = n % DHn;
          outH[(((size_t)b * Hn + h) * S + s) * DHn + dh] = (__bf16)val;
        }
      }
    }
  }
}

// ---------------------------------------------------------------------------
// Flash attention, causal + padding mask.
// Block = (qblk, h, b): 64 q rows, 128 threads = 4 waves; wave owns 16 q cols.
// S^T = K*Q^T (rows=keys, cols=q) so key reductions are lane-local; then
// O^T = V^T * P^T.  Q/K/V bf16 in [B,H,S,DH]; output bf16 in [B,S,H,DH].
// ---------------------------------------------------------------------------
#define VT_PITCH 72  // 64 + 8 pad: breaks the 32-dword stride bank collision

__global__ __launch_bounds__(128)
void k_attn(const __bf16* __restrict__ Qg, const __bf16* __restrict__ Kg,
            const __bf16* __restrict__ Vg, const int* __restrict__ pad,
            __bf16* __restrict__ Ob, int Hn, int S) {
  __shared__ __align__(16) __bf16 Ks[64][64];          // [key][dh]
  __shared__ __align__(16) __bf16 Vts[64][VT_PITCH];   // [dh][key] (transposed)
  __shared__ __align__(16) __bf16 Pw[4][16][64];       // per-wave P^T [q][key]
  __shared__ float mb[64];                             // padding bias per key

  const int qblk = blockIdx.x, h = blockIdx.y, b = blockIdx.z;
  const int tid  = threadIdx.x;
  const int wid  = tid >> 5;
  const int lane = tid & 31;
  const int half = lane >> 4;
  const int l16  = lane & 15;
  const int q    = qblk * 64 + wid * 16 + l16;  // this lane's q column

  // Q^T fragment for this lane's q column (dh is the WMMA K dim).
  const __bf16* qrow = Qg + (((size_t)b * Hn + h) * S + q) * 64;
  FragBF qf[2];
  qf[0].h[0] = *reinterpret_cast<const bf16x8*>(&qrow[half * 8]);
  qf[0].h[1] = *reinterpret_cast<const bf16x8*>(&qrow[16 + half * 8]);
  qf[1].h[0] = *reinterpret_cast<const bf16x8*>(&qrow[32 + half * 8]);
  qf[1].h[1] = *reinterpret_cast<const bf16x8*>(&qrow[48 + half * 8]);

  const __bf16* kbase = Kg + (((size_t)b * Hn + h) * S) * 64;
  const __bf16* vbase = Vg + (((size_t)b * Hn + h) * S) * 64;

  f32x8 accO[4];
  const f32x8 vzero = {};
#pragma unroll
  for (int dt = 0; dt < 4; ++dt) accO[dt] = vzero;
  float mrun = -1e30f, lrun = 0.f;
  const float inv_scale = 0.125f;  // 1/sqrt(DH) = 1/8

  const int r = tid >> 1, c = tid & 1;  // 64 rows x 2 segments of 32 bf16

  for (int kb = 0; kb <= qblk; ++kb) {
    __syncthreads();  // previous-iteration LDS reads done before overwrite
    // K tile: async straight into LDS.
#pragma unroll
    for (int i = 0; i < 4; ++i)
      cp_lds16(&Ks[r][c * 32 + i * 8],
               &kbase[((size_t)kb * 64 + r) * 64 + c * 32 + i * 8]);
    // V tile: through registers (needs transpose on the way into LDS).
    {
      i32x4 vtmp[4];
      const i32x4* gv = reinterpret_cast<const i32x4*>(
          &vbase[((size_t)kb * 64 + r) * 64 + c * 32]);
#pragma unroll
      for (int i = 0; i < 4; ++i) vtmp[i] = gv[i];
      const __bf16* tv = reinterpret_cast<const __bf16*>(vtmp);
#pragma unroll
      for (int j = 0; j < 32; ++j) Vts[c * 32 + j][r] = tv[j];
    }
    if (tid < 64) mb[tid] = (pad[(size_t)b * S + kb * 64 + tid] != 0) ? 0.f : -1e30f;
    cp_lds_wait<0>();
    __syncthreads();

    // ---- S^T tiles: rows = keys (4 tiles of 16), cols = q, K-dim = dh (64)
    f32x8 sc[4];
#pragma unroll
    for (int mt = 0; mt < 4; ++mt) {
      FragBF kf0, kf1;
      const int kr = mt * 16 + l16;
      kf0.h[0] = *reinterpret_cast<const bf16x8*>(&Ks[kr][half * 8]);
      kf0.h[1] = *reinterpret_cast<const bf16x8*>(&Ks[kr][16 + half * 8]);
      kf1.h[0] = *reinterpret_cast<const bf16x8*>(&Ks[kr][32 + half * 8]);
      kf1.h[1] = *reinterpret_cast<const bf16x8*>(&Ks[kr][48 + half * 8]);
      f32x8 z = vzero;
      z = WMMA_BF16(kf0.v, qf[0].v, z);
      sc[mt] = WMMA_BF16(kf1.v, qf[1].v, z);
    }

    // ---- scale + causal/padding mask + online softmax (lane-local keys)
    float mblk = -1e30f;
#pragma unroll
    for (int mt = 0; mt < 4; ++mt) {
#pragma unroll
      for (int e = 0; e < 8; ++e) {
        const int kl = mt * 16 + e + half * 8;
        const int kg = kb * 64 + kl;
        float sv = sc[mt][e] * inv_scale + mb[kl];
        if (kg > q) sv = -1e30f;
        sc[mt][e] = sv;
        mblk = fmaxf(mblk, sv);
      }
    }
    mblk = fmaxf(mblk, __shfl_xor(mblk, 16, 32));  // combine half-wave row groups
    const float mnew  = fmaxf(mrun, mblk);
    const float alpha = __expf(mrun - mnew);
    float psum = 0.f;
#pragma unroll
    for (int mt = 0; mt < 4; ++mt) {
#pragma unroll
      for (int e = 0; e < 8; ++e) {
        const float p = __expf(sc[mt][e] - mnew);
        sc[mt][e] = p;
        psum += p;
      }
    }
    psum += __shfl_xor(psum, 16, 32);
    lrun = lrun * alpha + psum;
    mrun = mnew;
#pragma unroll
    for (int dt = 0; dt < 4; ++dt)
#pragma unroll
      for (int e = 0; e < 8; ++e) accO[dt][e] *= alpha;

    // ---- P^T to LDS (bf16), per-wave region, keys contiguous per lane
#pragma unroll
    for (int mt = 0; mt < 4; ++mt) {
      bf16x8 pv;
#pragma unroll
      for (int e = 0; e < 8; ++e) pv[e] = (__bf16)sc[mt][e];
      *reinterpret_cast<bf16x8*>(&Pw[wid][l16][mt * 16 + half * 8]) = pv;
    }
    asm volatile("s_wait_dscnt 0" ::: "memory");  // same-wave LDS RAW

    // ---- O^T += V^T * P^T  (rows = dh, cols = q, K-dim = keys 64)
#pragma unroll
    for (int ks = 0; ks < 64; ks += 32) {
      FragBF pf;
      pf.h[0] = *reinterpret_cast<const bf16x8*>(&Pw[wid][l16][ks + half * 8]);
      pf.h[1] = *reinterpret_cast<const bf16x8*>(&Pw[wid][l16][ks + 16 + half * 8]);
#pragma unroll
      for (int dt = 0; dt < 4; ++dt) {
        FragBF vf;
        const int dr = dt * 16 + l16;
        vf.h[0] = *reinterpret_cast<const bf16x8*>(&Vts[dr][ks + half * 8]);
        vf.h[1] = *reinterpret_cast<const bf16x8*>(&Vts[dr][ks + 16 + half * 8]);
        accO[dt] = WMMA_BF16(vf.v, pf.v, accO[dt]);
      }
    }
  }

  // ---- normalize and write attended in [B, S, H, DH] bf16
  const float invl = (lrun > 0.f) ? (1.f / lrun) : 0.f;
#pragma unroll
  for (int dt = 0; dt < 4; ++dt) {
    bf16x8 ov;
#pragma unroll
    for (int e = 0; e < 8; ++e) ov[e] = (__bf16)(accO[dt][e] * invl);
    __bf16* orow = Ob + (((size_t)b * S + q) * Hn + h) * 64 + dt * 16 + half * 8;
    *reinterpret_cast<bf16x8*>(orow) = ov;
  }
}

// ---------------------------------------------------------------------------
extern "C" void kernel_launch(void* const* d_in, const int* in_sizes, int n_in,
                              void* d_out, int out_size, void* d_ws, size_t ws_size,
                              hipStream_t stream) {
  const float* x  = (const float*)d_in[0];
  const float* Wq = (const float*)d_in[1];
  const float* bq = (const float*)d_in[2];
  const float* Wk = (const float*)d_in[3];
  const float* bk = (const float*)d_in[4];
  const float* Wv = (const float*)d_in[5];
  const float* bv = (const float*)d_in[6];
  const float* Wo = (const float*)d_in[7];
  const float* bo = (const float*)d_in[8];
  const int*  pad = (const int*)d_in[9];
  float* out = (float*)d_out;

  const int Bc = 4, S = 2048, D = 1024, Hn = 16, DHn = 64;
  const int M = Bc * S;  // 8192 token rows

  char* w = (char*)d_ws;
  __bf16* xb   = (__bf16*)w; w += (size_t)M * D * 2;
  __bf16* wqt  = (__bf16*)w; w += (size_t)D * D * 2;
  __bf16* wkt  = (__bf16*)w; w += (size_t)D * D * 2;
  __bf16* wvt  = (__bf16*)w; w += (size_t)D * D * 2;
  __bf16* wot  = (__bf16*)w; w += (size_t)D * D * 2;
  __bf16* qb   = (__bf16*)w; w += (size_t)M * D * 2;
  __bf16* kbuf = (__bf16*)w; w += (size_t)M * D * 2;
  __bf16* vbuf = (__bf16*)w; w += (size_t)M * D * 2;
  __bf16* ab   = (__bf16*)w; w += (size_t)M * D * 2;
  (void)ws_size; (void)in_sizes; (void)n_in; (void)out_size;

  // 1) convert activations to bf16
  k_cvt_bf16<<<(M * D / 4 + 255) / 256, 256, 0, stream>>>(x, xb, M * D / 4);

  // 2) transpose+convert weights
  dim3 tb(32, 8), tg(D / 32, D / 32);
  k_transpose_bf16<<<tg, tb, 0, stream>>>(Wq, wqt, D, D);
  k_transpose_bf16<<<tg, tb, 0, stream>>>(Wk, wkt, D, D);
  k_transpose_bf16<<<tg, tb, 0, stream>>>(Wv, wvt, D, D);
  k_transpose_bf16<<<tg, tb, 0, stream>>>(Wo, wot, D, D);

  // 3) QKV projections (bf16 WMMA, fp32 accumulate, head-split output)
  dim3 gg(M / 128, D / 128);
  k_gemm_bf16<<<gg, 256, 0, stream>>>(xb, wqt, bq, nullptr, qb,   M, D, D, 1, S, Hn, DHn);
  k_gemm_bf16<<<gg, 256, 0, stream>>>(xb, wkt, bk, nullptr, kbuf, M, D, D, 1, S, Hn, DHn);
  k_gemm_bf16<<<gg, 256, 0, stream>>>(xb, wvt, bv, nullptr, vbuf, M, D, D, 1, S, Hn, DHn);

  // 4) causal flash attention
  dim3 ag(S / 64, Hn, Bc);
  k_attn<<<ag, 128, 0, stream>>>(qb, kbuf, vbuf, pad, ab, Hn, S);

  // 5) output projection -> fp32
  k_gemm_bf16<<<gg, 256, 0, stream>>>(ab, wot, bo, out, nullptr, M, D, D, 0, S, Hn, DHn);
}
